// ViSNetBlock_21217138442316
// MI455X (gfx1250) — compile-verified, matrix operand order checked by reading, not run on hardware
//
#include <hip/hip_runtime.h>
#include <hip/hip_bf16.h>

// ---------------- problem constants ----------------
#define N_NODES 8000
#define N_EDGES 160000
#define H 128
#define PI_F 3.14159265358979f
#define CUT_UPPER 5.0f

typedef __attribute__((ext_vector_type(16))) __bf16 v16bf;
typedef __attribute__((ext_vector_type(8)))  __bf16 v8bf;
typedef __attribute__((ext_vector_type(8)))  float  v8f;
typedef int v4i_t __attribute__((vector_size(16)));

#if defined(__has_builtin)
#if __has_builtin(__builtin_amdgcn_global_load_async_to_lds_b128) && \
    __has_builtin(__builtin_amdgcn_s_wait_asynccnt)
#define USE_ASYNC_LDS 1
#endif
#endif

#define GAS __attribute__((address_space(1)))
#define LAS __attribute__((address_space(3)))

__device__ __forceinline__ float silu_f(float x) {
  // x * sigmoid(x) with v_rcp_f32 instead of a full IEEE divide
  return x * __builtin_amdgcn_rcpf(1.0f + __expf(-x));
}

// Build a 16xbf16 WMMA fragment from two contiguous 8-element chunks.
// (CDNA5 16-bit A/B layout: lane l -> row/col (l&15), K-chunks at
//  (l>>4)*8 and (l>>4)*8+16, each chunk contiguous in memory.)
__device__ __forceinline__ v16bf ld_frag(const __bf16* p0, const __bf16* p1) {
  v8bf a = *(const v8bf*)p0;
  v8bf b = *(const v8bf*)p1;
  v16bf r;
#pragma unroll
  for (int i = 0; i < 8; ++i) { r[i] = a[i]; r[8 + i] = b[i]; }
  return r;
}

// ---------------- WMMA GEMM ----------------
// C[M x Nout] = act(A[M x 128] * W[Nout x 128]^T + bias)
// Block = 8 waves = 128-row M-supertile. A fragments live in registers for
// the whole N loop; W is streamed through LDS in 64-column groups staged
// with async global->LDS copies and shared by all 8 waves.
__global__ __launch_bounds__(256) void gemm_bf16_wmma_kernel(
    const __bf16* __restrict__ A, const __bf16* __restrict__ W,
    const float* __restrict__ bias, float* __restrict__ C,
    int M, int Nout, int act) {
  __shared__ __bf16 Bsh[64 * H];  // 16 KB weight group

  const int wave = threadIdx.x >> 5;
  const int lane = threadIdx.x & 31;
  const int rowBase = blockIdx.x * 128 + wave * 16;
  const bool active = (rowBase < M);  // wave-uniform

  const int frow = lane & 15;         // fragment row/col within tile
  const int ksel = (lane >> 4) << 3;  // 0 or 8

  // ---- load all A k-fragments once (kept in registers for all N groups)
  v16bf afr[4];
  if (active) {
    const __bf16* Ap = A + (size_t)(rowBase + frow) * H + ksel;
    __builtin_prefetch(Ap + (size_t)128 * H, 0, 1);  // next M-supertile
#pragma unroll
    for (int s = 0; s < 4; ++s)
      afr[s] = ld_frag(Ap + 32 * s, Ap + 32 * s + 16);
  }

  const int nGroups = Nout >> 6;  // 64-wide column groups
  for (int g = 0; g < nGroups; ++g) {
    // ---- cooperative stage: 64 rows of W (16 KB) into LDS
    const __bf16* Wg = W + (size_t)(g * 64) * H;
#pragma unroll
    for (int t = threadIdx.x; t < 1024; t += 256) {  // 1024 x 16B chunks
#ifdef USE_ASYNC_LDS
      __builtin_amdgcn_global_load_async_to_lds_b128(
          (GAS v4i_t*)(Wg + 8 * t), (LAS v4i_t*)(Bsh + 8 * t), 0, 0);
#else
      *(v8bf*)(Bsh + 8 * t) = *(const v8bf*)(Wg + 8 * t);
#endif
    }
#ifdef USE_ASYNC_LDS
    __builtin_amdgcn_s_wait_asynccnt(0);
#endif
    __syncthreads();

    if (active) {
#pragma unroll
      for (int t = 0; t < 4; ++t) {  // four 16-col tiles per group
        v8f c = {};
#pragma unroll
        for (int s = 0; s < 4; ++s) {
          const __bf16* bp = &Bsh[(t * 16 + frow) * H + 32 * s + ksel];
          v16bf b = ld_frag(bp, bp + 16);  // ds_load_b128 x2
          c = __builtin_amdgcn_wmma_f32_16x16x32_bf16(false, afr[s], false, b,
                                                      (short)0, c, false,
                                                      false);
        }
        const int ncol = g * 64 + t * 16 + (lane & 15);
        const float bi = bias ? bias[ncol] : 0.0f;
        const int row0 = rowBase + ((lane >> 4) << 3);
#pragma unroll
        for (int r = 0; r < 8; ++r) {
          float y = c[r] + bi;
          if (act) y = silu_f(y);
          C[(size_t)(row0 + r) * Nout + ncol] = y;
        }
      }
    }
    __syncthreads();  // protect Bsh before next group's staging
  }
}

// ---------------- f32 -> bf16 conversion ----------------
__global__ void f2bf_kernel(const float* __restrict__ src,
                            __bf16* __restrict__ dst, int n) {
  int i = blockIdx.x * 256 + threadIdx.x;
  if (i < n) dst[i] = (__bf16)src[i];
}

// ---------------- LayerNorm over rows of length H, bf16 output ----------------
__global__ __launch_bounds__(H) void ln_rows_kernel(
    const float* __restrict__ X, const float* __restrict__ w,
    const float* __restrict__ b, __bf16* __restrict__ out) {
  const int row = blockIdx.x;
  const int h = threadIdx.x;
  float v = X[(size_t)row * H + h];
  float s = v, s2 = v * v;
#pragma unroll
  for (int o = 16; o > 0; o >>= 1) {
    s += __shfl_down(s, o, 32);
    s2 += __shfl_down(s2, o, 32);
  }
  __shared__ float sh[8];
  const int wid = h >> 5, lane = h & 31;
  if (lane == 0) { sh[wid] = s; sh[4 + wid] = s2; }
  __syncthreads();
  const float ts = sh[0] + sh[1] + sh[2] + sh[3];
  const float ts2 = sh[4] + sh[5] + sh[6] + sh[7];
  const float mu = ts * (1.0f / H);
  const float var = ts2 * (1.0f / H) - mu * mu;
  const float y = (v - mu) * rsqrtf(var + 1e-5f) * w[h] + b[h];
  out[(size_t)row * H + h] = (__bf16)y;
}

// ---------------- vector LayerNorm (equivariant), f32 + bf16 outputs ----------------
__global__ __launch_bounds__(H) void vec_ln_kernel(
    const float* __restrict__ vec, const float* __restrict__ w,
    float* __restrict__ vn, __bf16* __restrict__ vn_bf) {
  const int n = blockIdx.x;
  const int h = threadIdx.x;
  const float v0 = vec[((size_t)n * 3 + 0) * H + h];
  const float v1 = vec[((size_t)n * 3 + 1) * H + h];
  const float v2 = vec[((size_t)n * 3 + 2) * H + h];
  const float dist = sqrtf(v0 * v0 + v1 * v1 + v2 * v2);
  const float distc = fmaxf(dist, 1e-12f);
  float mx = distc, mn = distc;
#pragma unroll
  for (int o = 16; o > 0; o >>= 1) {
    mx = fmaxf(mx, __shfl_xor(mx, o, 32));
    mn = fminf(mn, __shfl_xor(mn, o, 32));
  }
  __shared__ float shx[4], shn[4];
  const int wid = h >> 5, lane = h & 31;
  if (lane == 0) { shx[wid] = mx; shn[wid] = mn; }
  __syncthreads();
  const float MX = fmaxf(fmaxf(shx[0], shx[1]), fmaxf(shx[2], shx[3]));
  const float MN = fminf(fminf(shn[0], shn[1]), fminf(shn[2], shn[3]));
  float delta = MX - MN;
  if (delta == 0.0f) delta = 1.0f;
  const float distn = fmaxf((distc - MN) * __builtin_amdgcn_rcpf(delta), 0.0f);
  const float sc = distn * __builtin_amdgcn_rcpf(distc) * w[h];
  const float o0 = v0 * sc, o1 = v1 * sc, o2 = v2 * sc;
  vn[((size_t)n * 3 + 0) * H + h] = o0;
  vn[((size_t)n * 3 + 1) * H + h] = o1;
  vn[((size_t)n * 3 + 2) * H + h] = o2;
  vn_bf[((size_t)n * 3 + 0) * H + h] = (__bf16)o0;
  vn_bf[((size_t)n * 3 + 1) * H + h] = (__bf16)o1;
  vn_bf[((size_t)n * 3 + 2) * H + h] = (__bf16)o2;
}

// ---------------- vec_dot input: sum_c vp[n,c,h]*vp[n,c,H+h] ----------------
__global__ __launch_bounds__(H) void vec_dot_in_kernel(
    const float* __restrict__ vp, __bf16* __restrict__ out) {
  const int n = blockIdx.x;
  const int h = threadIdx.x;
  float acc = 0.0f;
#pragma unroll
  for (int c = 0; c < 3; ++c) {
    const float a = vp[((size_t)n * 3 + c) * 384 + h];
    const float b = vp[((size_t)n * 3 + c) * 384 + H + h];
    acc += a * b;
  }
  out[(size_t)n * H + h] = (__bf16)acc;
}

// ---------------- fused edge attention + vj + x_agg scatter ----------------
__global__ __launch_bounds__(H) void edge_attn_vj_kernel(
    const float* __restrict__ q, const float* __restrict__ k,
    const float* __restrict__ v, const float* __restrict__ dk,
    const float* __restrict__ dv, const int* __restrict__ ei,
    const float* __restrict__ r_ij, float* __restrict__ x_agg,
    __bf16* __restrict__ vj_bf) {
  const int e = blockIdx.x;
  const int h = threadIdx.x;
  const int src = ei[e];
  const int dst = ei[N_EDGES + e];
  float p = q[(size_t)dst * H + h] * k[(size_t)src * H + h] *
            dk[(size_t)e * H + h];
#pragma unroll
  for (int o = 8; o > 0; o >>= 1) p += __shfl_xor(p, o, 16);  // head dim = 16
  const float r = r_ij[e];
  const float cut =
      (r < CUT_UPPER) ? 0.5f * (__cosf(r * (PI_F / CUT_UPPER)) + 1.0f) : 0.0f;
  const float at = silu_f(p) * cut;
  const float vj = v[(size_t)src * H + h] * dv[(size_t)e * H + h] * at;
  atomicAdd(&x_agg[(size_t)dst * H + h], vj);
  vj_bf[(size_t)e * H + h] = (__bf16)vj;
}

// ---------------- vec_msg scatter: vn[src]*s1 + s2*d_ij ----------------
__global__ __launch_bounds__(H) void edge_vecmsg_kernel(
    const float* __restrict__ vn, const float* __restrict__ s,
    const float* __restrict__ d_ij, const int* __restrict__ ei,
    float* __restrict__ vec_agg) {
  const int e = blockIdx.x;
  const int h = threadIdx.x;
  const int src = ei[e];
  const int dst = ei[N_EDGES + e];
  const float s1 = s[(size_t)e * 2 * H + h];
  const float s2 = s[(size_t)e * 2 * H + H + h];
#pragma unroll
  for (int c = 0; c < 3; ++c) {
    const float msg =
        vn[((size_t)src * 3 + c) * H + h] * s1 + s2 * d_ij[(size_t)e * 3 + c];
    atomicAdd(&vec_agg[((size_t)dst * 3 + c) * H + h], msg);
  }
}

// ---------------- node residual update ----------------
__global__ __launch_bounds__(H) void node_update_kernel(
    float* __restrict__ x_cur, float* __restrict__ vec_cur,
    const float* __restrict__ vec_dot, const float* __restrict__ o,
    const float* __restrict__ vp, const float* __restrict__ vec_agg) {
  const int n = blockIdx.x;
  const int h = threadIdx.x;
  const float o1 = o[(size_t)n * 384 + h];
  const float o2 = o[(size_t)n * 384 + H + h];
  const float o3 = o[(size_t)n * 384 + 2 * H + h];
  x_cur[(size_t)n * H + h] += vec_dot[(size_t)n * H + h] * o2 + o3;
#pragma unroll
  for (int c = 0; c < 3; ++c) {
    const float v3 = vp[((size_t)n * 3 + c) * 384 + 2 * H + h];  // vec3
    vec_cur[((size_t)n * 3 + c) * H + h] +=
        v3 * o1 + vec_agg[((size_t)n * 3 + c) * H + h];
  }
}

// ---------------- w_dot input: <vrej(wtrg[dst],d), vrej(wsrc[src],-d)> ----------------
__global__ __launch_bounds__(H) void edge_wdot_in_kernel(
    const float* __restrict__ wtrg, const float* __restrict__ wsrc,
    const float* __restrict__ d_ij, const int* __restrict__ ei,
    __bf16* __restrict__ out) {
  const int e = blockIdx.x;
  const int h = threadIdx.x;
  const int src = ei[e];
  const int dst = ei[N_EDGES + e];
  const float d0 = d_ij[(size_t)e * 3 + 0];
  const float d1 = d_ij[(size_t)e * 3 + 1];
  const float d2 = d_ij[(size_t)e * 3 + 2];
  const float g10 = wtrg[((size_t)dst * 3 + 0) * H + h];
  const float g11 = wtrg[((size_t)dst * 3 + 1) * H + h];
  const float g12 = wtrg[((size_t)dst * 3 + 2) * H + h];
  const float g20 = wsrc[((size_t)src * 3 + 0) * H + h];
  const float g21 = wsrc[((size_t)src * 3 + 1) * H + h];
  const float g22 = wsrc[((size_t)src * 3 + 2) * H + h];
  const float p1 = g10 * d0 + g11 * d1 + g12 * d2;
  const float p2 = g20 * d0 + g21 * d1 + g22 * d2;  // vrej(v,-d) == vrej(v,d)
  const float w10 = g10 - p1 * d0, w11 = g11 - p1 * d1, w12 = g12 - p1 * d2;
  const float w20 = g20 - p2 * d0, w21 = g21 - p2 * d1, w22 = g22 - p2 * d2;
  out[(size_t)e * H + h] = (__bf16)(w10 * w20 + w11 * w21 + w12 * w22);
}

// ---------------- edge feature residual: f += ff1*w_dot + ff2 ----------------
__global__ __launch_bounds__(H) void f_update_kernel(
    float* __restrict__ f_cur, const float* __restrict__ ff,
    const float* __restrict__ w_dot) {
  const int e = blockIdx.x;
  const int h = threadIdx.x;
  f_cur[(size_t)e * H + h] +=
      ff[(size_t)e * 2 * H + h] * w_dot[(size_t)e * H + h] +
      ff[(size_t)e * 2 * H + H + h];
}

// =====================================================================
extern "C" void kernel_launch(void* const* d_in, const int* in_sizes, int n_in,
                              void* d_out, int out_size, void* d_ws,
                              size_t ws_size, hipStream_t stream) {
  const int N = N_NODES, E = N_EDGES;

  // -------- unpack inputs --------
  const float* x_in = (const float*)d_in[0];
  const float* vec_in = (const float*)d_in[1];
  const int* ei = (const int*)d_in[2];
  const float* r_ij = (const float*)d_in[3];
  const float* f_in = (const float*)d_in[4];
  const float* d_ij = (const float*)d_in[5];

  // params: layer dicts flattened in insertion order
  const float* P[3][28] = {};
  {
    int idx = 6;
    for (int l = 0; l < 3; ++l) {
      int cnt = (l < 2) ? 28 : 22;
      for (int j = 0; j < cnt; ++j) P[l][j] = (const float*)d_in[idx++];
    }
  }
  enum { LNXW=0, LNXB, LNFW, LNFB, VLNW, WQ, BQ, WK, BK, WV, BV, WS, BS,
         WVD, BVD, WO, BO, WVEC, WDK, BDK, WDV, BDV, WF, BF_, WSRC, WTRG,
         WWD, BWD };

  // -------- workspace bump allocator --------
  size_t off = 0;
  auto alloc = [&](size_t bytes) -> void* {
    void* p = (char*)d_ws + off;
    off += (bytes + 255) & ~(size_t)255;
    return p;
  };
  float* x_cur = (float*)alloc((size_t)N * H * 4);
  float* vec_cur = (float*)alloc((size_t)N * 3 * H * 4);
  float* f_cur = (float*)alloc((size_t)E * H * 4);
  __bf16* xn_bf = (__bf16*)alloc((size_t)N * H * 2);
  __bf16* fn_bf = (__bf16*)alloc((size_t)E * H * 2);
  float* vn = (float*)alloc((size_t)N * 3 * H * 4);
  __bf16* vn_bf = (__bf16*)alloc((size_t)N * 3 * H * 2);
  float* qb = (float*)alloc((size_t)N * H * 4);
  float* kb = (float*)alloc((size_t)N * H * 4);
  float* vb = (float*)alloc((size_t)N * H * 4);
  float* vp = (float*)alloc((size_t)N * 3 * 384 * 4);
  __bf16* vdin_bf = (__bf16*)alloc((size_t)N * H * 2);
  float* vec_dot = (float*)alloc((size_t)N * H * 4);
  float* dk = (float*)alloc((size_t)E * H * 4);   // reused as w_dot later
  float* dv = (float*)alloc((size_t)E * H * 4);
  __bf16* vj_bf = (__bf16*)alloc((size_t)E * H * 2);
  float* s_buf = (float*)alloc((size_t)E * 2 * H * 4);  // reused as ff later
  float* x_agg = (float*)alloc((size_t)N * H * 4);
  __bf16* xagg_bf = (__bf16*)alloc((size_t)N * H * 2);
  float* vec_agg = (float*)alloc((size_t)N * 3 * H * 4);
  float* o_buf = (float*)alloc((size_t)N * 384 * 4);
  float* wtrg = (float*)alloc((size_t)N * 3 * H * 4);
  float* wsrc = (float*)alloc((size_t)N * 3 * H * 4);
  __bf16* wdin_bf = (__bf16*)alloc((size_t)E * H * 2);
  __bf16* warena = (__bf16*)alloc((size_t)400 * 1024 * 2);  // bf16 weights
  float* w_dot = dk;  // safe reuse: dk consumed before w_dot produced

  // -------- helpers --------
  auto cvt = [&](const float* src, __bf16* dst, size_t n) {
    f2bf_kernel<<<(unsigned)((n + 255) / 256), 256, 0, stream>>>(src, dst,
                                                                 (int)n);
  };
  auto gemm = [&](const __bf16* A, const __bf16* W, const float* bias,
                  float* C, int M, int Nout, int act) {
    unsigned blocks = (unsigned)((M + 127) / 128);
    gemm_bf16_wmma_kernel<<<blocks, 256, 0, stream>>>(A, W, bias, C, M, Nout,
                                                      act);
  };

  // -------- init running state --------
  (void)hipMemcpyAsync(x_cur, x_in, (size_t)N * H * 4,
                       hipMemcpyDeviceToDevice, stream);
  (void)hipMemcpyAsync(vec_cur, vec_in, (size_t)N * 3 * H * 4,
                       hipMemcpyDeviceToDevice, stream);
  (void)hipMemcpyAsync(f_cur, f_in, (size_t)E * H * 4,
                       hipMemcpyDeviceToDevice, stream);

  // -------- 3 layers --------
  for (int l = 0; l < 3; ++l) {
    const bool last = (l == 2);
    const float* const* p = P[l];

    // bf16 weight arena for this layer
    size_t wo = 0;
    auto wal = [&](size_t n) { __bf16* q = warena + wo; wo += n; return q; };
    __bf16* Wq_bf = wal(H * H);   cvt(p[WQ], Wq_bf, H * H);
    __bf16* Wk_bf = wal(H * H);   cvt(p[WK], Wk_bf, H * H);
    __bf16* Wv_bf = wal(H * H);   cvt(p[WV], Wv_bf, H * H);
    __bf16* Wvd_bf = wal(H * H);  cvt(p[WVD], Wvd_bf, H * H);
    __bf16* Wdk_bf = wal(H * H);  cvt(p[WDK], Wdk_bf, H * H);
    __bf16* Wdv_bf = wal(H * H);  cvt(p[WDV], Wdv_bf, H * H);
    __bf16* Ws_bf = wal(2 * H * H);  cvt(p[WS], Ws_bf, 2 * H * H);
    __bf16* Wo_bf = wal(3 * H * H);  cvt(p[WO], Wo_bf, 3 * H * H);
    __bf16* Wvec_bf = wal(3 * H * H); cvt(p[WVEC], Wvec_bf, 3 * H * H);
    __bf16 *Wf_bf = nullptr, *Wsrc_bf = nullptr, *Wtrg_bf = nullptr,
           *Wwd_bf = nullptr;
    if (!last) {
      Wf_bf = wal(2 * H * H);   cvt(p[WF], Wf_bf, 2 * H * H);
      Wsrc_bf = wal(H * H);     cvt(p[WSRC], Wsrc_bf, H * H);
      Wtrg_bf = wal(H * H);     cvt(p[WTRG], Wtrg_bf, H * H);
      Wwd_bf = wal(H * H);      cvt(p[WWD], Wwd_bf, H * H);
    }

    // norms
    ln_rows_kernel<<<N, H, 0, stream>>>(x_cur, p[LNXW], p[LNXB], xn_bf);
    ln_rows_kernel<<<E, H, 0, stream>>>(f_cur, p[LNFW], p[LNFB], fn_bf);
    vec_ln_kernel<<<N, H, 0, stream>>>(vec_cur, p[VLNW], vn, vn_bf);

    // projections (WMMA)
    gemm(xn_bf, Wq_bf, p[BQ], qb, N, H, 0);
    gemm(xn_bf, Wk_bf, p[BK], kb, N, H, 0);
    gemm(xn_bf, Wv_bf, p[BV], vb, N, H, 0);
    gemm(vn_bf, Wvec_bf, nullptr, vp, N * 3, 3 * H, 0);
    gemm(fn_bf, Wdk_bf, p[BDK], dk, E, H, 1);
    gemm(fn_bf, Wdv_bf, p[BDV], dv, E, H, 1);

    // vec_dot = silu((vec1*vec2).sum(c) @ Wvd^T + bvd)
    vec_dot_in_kernel<<<N, H, 0, stream>>>(vp, vdin_bf);
    gemm(vdin_bf, Wvd_bf, p[BVD], vec_dot, N, H, 1);

    // aggregation buffers
    (void)hipMemsetAsync(x_agg, 0, (size_t)N * H * 4, stream);
    (void)hipMemsetAsync(vec_agg, 0, (size_t)N * 3 * H * 4, stream);

    // edge attention (fused gather, head-reduce, cutoff, vj, x scatter)
    edge_attn_vj_kernel<<<E, H, 0, stream>>>(qb, kb, vb, dk, dv, ei, r_ij,
                                             x_agg, vj_bf);

    // s = silu(vj @ Ws^T + bs); vec_msg scatter
    gemm(vj_bf, Ws_bf, p[BS], s_buf, E, 2 * H, 1);
    edge_vecmsg_kernel<<<E, H, 0, stream>>>(vn, s_buf, d_ij, ei, vec_agg);

    // o = x_agg @ Wo^T + bo ; residual updates
    cvt(x_agg, xagg_bf, (size_t)N * H);
    gemm(xagg_bf, Wo_bf, p[BO], o_buf, N, 3 * H, 0);
    node_update_kernel<<<N, H, 0, stream>>>(x_cur, vec_cur, vec_dot, o_buf,
                                            vp, vec_agg);

    if (!last) {
      // directional gating for edge features
      gemm(vn_bf, Wtrg_bf, nullptr, wtrg, N * 3, H, 0);
      gemm(vn_bf, Wsrc_bf, nullptr, wsrc, N * 3, H, 0);
      edge_wdot_in_kernel<<<E, H, 0, stream>>>(wtrg, wsrc, d_ij, ei, wdin_bf);
      gemm(wdin_bf, Wwd_bf, p[BWD], w_dot, E, H, 1);     // w_dot reuses dk
      gemm(fn_bf, Wf_bf, p[BF_], s_buf, E, 2 * H, 1);    // ff reuses s_buf
      f_update_kernel<<<E, H, 0, stream>>>(f_cur, s_buf, w_dot);
    }
  }

  // -------- outputs: concat(x, vec) --------
  float* out_f = (float*)d_out;
  (void)hipMemcpyAsync(out_f, x_cur, (size_t)N * H * 4,
                       hipMemcpyDeviceToDevice, stream);
  (void)hipMemcpyAsync(out_f + (size_t)N * H, vec_cur, (size_t)N * 3 * H * 4,
                       hipMemcpyDeviceToDevice, stream);
  (void)in_sizes; (void)n_in; (void)out_size; (void)ws_size;
}
